// KANEnhancedGNN_84619445666122
// MI455X (gfx1250) — matrix operand order repaired and checked by reading, not compiled, not run on hardware
//
#include <hip/hip_runtime.h>
#include <hip/hip_bf16.h>
#include <math.h>

// ---------------------------------------------------------------------------
// KAN-enhanced GNN forward for MI455X (gfx1250, wave32, WMMA).
// All matmuls lowered to v_wmma_f32_16x16x32_f16 (f16 operands, f32 accum).
// ---------------------------------------------------------------------------

#define NNODES   32768
#define NGRAPHS  64
#define NPER     512
#define INDIM    128
#define HID      256
#define OUTDIM   8
#define EDGEDIM  64
#define NEDGES   524288
#define HEADS    8
#define HDIM     32
#define ATTDIM   64
#define GRIDG    5
#define EPSK     1e-8f

typedef __attribute__((ext_vector_type(16))) _Float16 v16h;
typedef __attribute__((ext_vector_type(8)))  float    v8f;

__device__ __forceinline__ v8f wmma_f16(v16h a, v16h b, v8f c) {
  return __builtin_amdgcn_wmma_f32_16x16x32_f16(false, a, false, b, (short)0, c,
                                                false, false);
}

// A fragment (16x32 f16) from row-major LDS array (lda in halves).
// Lane layout per ISA 7.12.2: lanes 0-15 hold M=0..15 K{0..7,16..23},
// lanes 16-31 hold M=0..15 K{8..15,24..31}.
__device__ __forceinline__ v16h load_a_frag(const _Float16* As, int lda, int m0,
                                            int k0, int lane) {
  const _Float16* p = As + (m0 + (lane & 15)) * lda + k0 + ((lane >> 4) << 3);
  v16h a;
#pragma unroll
  for (int j = 0; j < 8; ++j) { a[j] = p[j]; a[j + 8] = p[j + 16]; }
  return a;
}

// B fragment (32x16 f16): Bs stored k-major [32][ldb]; lane%16 = N, K as A frag.
__device__ __forceinline__ v16h load_b_frag(const _Float16* Bs, int ldb, int n0,
                                            int lane) {
  const _Float16* p = Bs + ((lane >> 4) << 3) * ldb + n0 + (lane & 15);
  v16h b;
#pragma unroll
  for (int j = 0; j < 8; ++j) { b[j] = p[j * ldb]; b[j + 8] = p[(j + 16) * ldb]; }
  return b;
}

// Faithful port of the reference b_spline_basis recursion (G=5, k=3).
__device__ __forceinline__ void bspline5(float x, const float* g, float* b) {
  float d[5];
#pragma unroll
  for (int i = 0; i < 5; ++i) {
    d[i] = x - g[i];
    b[i] = (d[i] >= 0.f && d[i] < 1.f) ? 1.f : 0.f;
  }
#pragma unroll
  for (int order = 1; order <= 3; ++order) {
    const int L = 4 - order;
    float prev[5];
#pragma unroll
    for (int i = 0; i < 5; ++i) prev[i] = b[i];
#pragma unroll
    for (int j = 0; j < 4; ++j) {
      if (j < L) {
        float t1 = (d[j] - g[j]) / (g[j + order] - g[j] + EPSK) * prev[j];
        float t2 = (g[j + order + 1] - d[j]) /
                   (g[j + order + 1] - g[j + 1] + EPSK) * prev[j + 1];
        b[j] = t1 + t2;
      }
    }
  }
}

__device__ __forceinline__ float gelu_exact(float x) {
  return 0.5f * x * (1.f + erff(x * 0.70710678118654752f));
}

enum { EPI_NONE = 0, EPI_RELU = 1, EPI_GELU = 2 };

// ---------------------------------------------------------------------------
// Generic WMMA GEMM: C[M,N] = A[M,K] @ W[N,K]^T + bias, optional activation.
// 256 threads = 8 waves (4x2), block tile 128x64, K-step 32.
// ---------------------------------------------------------------------------
__global__ void gemm_wmma_kernel(const float* __restrict__ A,
                                 const float* __restrict__ W,
                                 const float* __restrict__ bias,
                                 float* __restrict__ C,
                                 int M, int N, int K, int epi) {
  extern __shared__ char smem_raw[];
  _Float16* As = (_Float16*)smem_raw;   // [128][32]
  _Float16* Bs = As + 128 * 32;         // [32][64] k-major
  const int tid = threadIdx.x, lane = tid & 31, wave = tid >> 5;
  const int wm = wave >> 1, wn = wave & 1;
  const int m0blk = blockIdx.x * 128, n0blk = blockIdx.y * 64;
  v8f acc[2][2] = {};

  for (int k0 = 0; k0 < K; k0 += 32) {
    __syncthreads();
#pragma unroll
    for (int t = 0; t < 4; ++t) {            // A: 128x32 f32 -> f16 LDS
      int idx = tid + t * 256;               // 1024 float4 slots
      int row = idx >> 3, c4 = (idx & 7) << 2;
      int gr = m0blk + row;
      float4 v = (gr < M) ? *(const float4*)(A + (size_t)gr * K + k0 + c4)
                          : make_float4(0.f, 0.f, 0.f, 0.f);
      _Float16* d = As + row * 32 + c4;
      d[0] = (_Float16)v.x; d[1] = (_Float16)v.y;
      d[2] = (_Float16)v.z; d[3] = (_Float16)v.w;
    }
#pragma unroll
    for (int t = 0; t < 2; ++t) {            // W: 64x32 -> Bs[k][n]
      int idx = tid + t * 256;               // 512 float4 slots
      int n = idx >> 3, c4 = (idx & 7) << 2;
      int gn = n0blk + n;
      float4 v = (gn < N) ? *(const float4*)(W + (size_t)gn * K + k0 + c4)
                          : make_float4(0.f, 0.f, 0.f, 0.f);
      Bs[(c4 + 0) * 64 + n] = (_Float16)v.x;
      Bs[(c4 + 1) * 64 + n] = (_Float16)v.y;
      Bs[(c4 + 2) * 64 + n] = (_Float16)v.z;
      Bs[(c4 + 3) * 64 + n] = (_Float16)v.w;
    }
    __syncthreads();
#pragma unroll
    for (int ni = 0; ni < 2; ++ni) {
      v16h bf = load_b_frag(Bs, 64, wn * 32 + ni * 16, lane);
#pragma unroll
      for (int mi = 0; mi < 2; ++mi) {
        v16h af = load_a_frag(As, 32, wm * 32 + mi * 16, 0, lane);
        acc[mi][ni] = wmma_f16(af, bf, acc[mi][ni]);
      }
    }
  }
  const int mo = (lane >> 4) << 3;
#pragma unroll
  for (int mi = 0; mi < 2; ++mi)
#pragma unroll
    for (int ni = 0; ni < 2; ++ni) {
      int n = n0blk + wn * 32 + ni * 16 + (lane & 15);
      float bv = (bias && n < N) ? bias[n] : 0.f;
#pragma unroll
      for (int r = 0; r < 8; ++r) {
        int m = m0blk + wm * 32 + mi * 16 + mo + r;
        if (m < M && n < N) {
          float v = acc[mi][ni][r] + bv;
          if (epi == EPI_RELU) v = fmaxf(v, 0.f);
          else if (epi == EPI_GELU) v = gelu_exact(v);
          C[(size_t)m * N + n] = v;
        }
      }
    }
}

// ---------------------------------------------------------------------------
// Edge KAN + gather + atomic segment-sum (one block = 64 edges, full N=256).
// msg[e] = kan_linear(edge_attr[e]) + h[src[e]]; agg[dst[e]] += msg (atomic).
// LDS: attr f16 [64][64] + spline basis f16 [64][320] + Bs f16 [32][256].
// ---------------------------------------------------------------------------
__global__ void kan_edge_agg_kernel(const float* __restrict__ edge_attr,
                                    const int* __restrict__ esrc,
                                    const int* __restrict__ edst,
                                    const float* __restrict__ h,
                                    const float* __restrict__ We,
                                    const float* __restrict__ Se,
                                    const float* __restrict__ be,
                                    const float* __restrict__ ge,
                                    float* __restrict__ agg) {
  extern __shared__ char smem_raw[];
  _Float16* attrS  = (_Float16*)smem_raw;   // 64*64
  _Float16* basisS = attrS + 64 * 64;       // 64*320
  _Float16* Bs     = basisS + 64 * 320;     // 32*256 k-major
  __shared__ float gS[EDGEDIM * GRIDG];
  const int tid = threadIdx.x, lane = tid & 31, w = tid >> 5;
  const int e0 = blockIdx.x * 64;

  for (int i = tid; i < EDGEDIM * GRIDG; i += 256) gS[i] = ge[i];
  __syncthreads();
#pragma unroll 4
  for (int t = 0; t < 16; ++t) {            // basis + f16 activations
    int idx = tid + t * 256;                // 64*64 pairs
    int e = idx >> 6, i = idx & 63;
    float x = edge_attr[(size_t)(e0 + e) * EDGEDIM + i];
    attrS[e * 64 + i] = (_Float16)x;
    float b5[GRIDG];
    bspline5(x, gS + i * GRIDG, b5);
    _Float16* bp = basisS + e * 320 + i * GRIDG;
#pragma unroll
    for (int j = 0; j < GRIDG; ++j) bp[j] = (_Float16)b5[j];
  }

  v8f acc[4][2] = {};
  for (int k0 = 0; k0 < 64; k0 += 32) {     // base term, K=64
    __syncthreads();
#pragma unroll
    for (int t = 0; t < 8; ++t) {
      int idx = tid + t * 256;
      int n = idx >> 3, c4 = (idx & 7) << 2;
      float4 v = *(const float4*)(We + (size_t)n * 64 + k0 + c4);
      Bs[(c4 + 0) * 256 + n] = (_Float16)v.x;
      Bs[(c4 + 1) * 256 + n] = (_Float16)v.y;
      Bs[(c4 + 2) * 256 + n] = (_Float16)v.z;
      Bs[(c4 + 3) * 256 + n] = (_Float16)v.w;
    }
    __syncthreads();
#pragma unroll
    for (int ni = 0; ni < 2; ++ni) {
      v16h bf = load_b_frag(Bs, 256, w * 32 + ni * 16, lane);
#pragma unroll
      for (int mi = 0; mi < 4; ++mi) {
        v16h af = load_a_frag(attrS, 64, mi * 16, k0, lane);
        acc[mi][ni] = wmma_f16(af, bf, acc[mi][ni]);
      }
    }
  }
  for (int k0 = 0; k0 < 320; k0 += 32) {    // spline term, K=64*5
    __syncthreads();
#pragma unroll
    for (int t = 0; t < 8; ++t) {
      int idx = tid + t * 256;
      int n = idx >> 3, c4 = (idx & 7) << 2;
      float4 v = *(const float4*)(Se + (size_t)n * 320 + k0 + c4);
      Bs[(c4 + 0) * 256 + n] = (_Float16)v.x;
      Bs[(c4 + 1) * 256 + n] = (_Float16)v.y;
      Bs[(c4 + 2) * 256 + n] = (_Float16)v.z;
      Bs[(c4 + 3) * 256 + n] = (_Float16)v.w;
    }
    __syncthreads();
#pragma unroll
    for (int ni = 0; ni < 2; ++ni) {
      v16h bf = load_b_frag(Bs, 256, w * 32 + ni * 16, lane);
#pragma unroll
      for (int mi = 0; mi < 4; ++mi) {
        v16h af = load_a_frag(basisS, 320, mi * 16, k0, lane);
        acc[mi][ni] = wmma_f16(af, bf, acc[mi][ni]);
      }
    }
  }
  const int mo = (lane >> 4) << 3;
#pragma unroll
  for (int mi = 0; mi < 4; ++mi)
#pragma unroll
    for (int ni = 0; ni < 2; ++ni) {
      int n = w * 32 + ni * 16 + (lane & 15);
      float bv = be[n];
#pragma unroll
      for (int r = 0; r < 8; ++r) {
        int e = e0 + mi * 16 + mo + r;
        int s = esrc[e], d = edst[e];
        float val = acc[mi][ni][r] + bv + h[(size_t)s * HID + n];
        atomicAdd(&agg[(size_t)d * HID + n], val);
      }
    }
}

// ---------------------------------------------------------------------------
// Node KAN + GELU + LayerNorm + scaled residual, fused (64 rows/block).
// Uses CDNA5's 320KB LDS: full 64x1280 f16 spline basis stays resident.
// ---------------------------------------------------------------------------
__global__ void kan_node_kernel(const float* __restrict__ agg,
                                const float* __restrict__ Wn,
                                const float* __restrict__ Sn,
                                const float* __restrict__ bn,
                                const float* __restrict__ gn,
                                const float* __restrict__ lng,
                                const float* __restrict__ lnb,
                                const float* __restrict__ resb,
                                const float* __restrict__ lam,
                                float* __restrict__ hout,
                                float* __restrict__ feat) {
  extern __shared__ char smem_raw[];
  _Float16* attrS  = (_Float16*)smem_raw;        // 64*256  (32 KB)
  _Float16* basisS = attrS + 64 * 256;           // 64*1280 (160 KB)
  _Float16* Bs     = basisS + 64 * 1280;         // 32*256  (16 KB)
  float*    outS   = (float*)(Bs + 32 * 256);    // 64*256  (64 KB)
  __shared__ float gS[HID * GRIDG];
  __shared__ float redS[64][4], redQ[64][4];
  const int tid = threadIdx.x, lane = tid & 31, w = tid >> 5;
  const size_t n0 = (size_t)blockIdx.x * 64;

  for (int i = tid; i < HID * GRIDG; i += 256) gS[i] = gn[i];
  __syncthreads();
  for (int t = 0; t < 64; ++t) {                 // 64*256 (row,feat) pairs
    int idx = tid + t * 256;
    int e = idx >> 8, i = idx & 255;
    float x = agg[(n0 + e) * HID + i];
    attrS[e * 256 + i] = (_Float16)x;
    float b5[GRIDG];
    bspline5(x, gS + i * GRIDG, b5);
    _Float16* bp = basisS + e * 1280 + i * GRIDG;
#pragma unroll
    for (int j = 0; j < GRIDG; ++j) bp[j] = (_Float16)b5[j];
  }

  v8f acc[4][2] = {};
  for (int k0 = 0; k0 < 256; k0 += 32) {         // base term, K=256
    __syncthreads();
#pragma unroll
    for (int t = 0; t < 8; ++t) {
      int idx = tid + t * 256;
      int n = idx >> 3, c4 = (idx & 7) << 2;
      float4 v = *(const float4*)(Wn + (size_t)n * 256 + k0 + c4);
      Bs[(c4 + 0) * 256 + n] = (_Float16)v.x;
      Bs[(c4 + 1) * 256 + n] = (_Float16)v.y;
      Bs[(c4 + 2) * 256 + n] = (_Float16)v.z;
      Bs[(c4 + 3) * 256 + n] = (_Float16)v.w;
    }
    __syncthreads();
#pragma unroll
    for (int ni = 0; ni < 2; ++ni) {
      v16h bf = load_b_frag(Bs, 256, w * 32 + ni * 16, lane);
#pragma unroll
      for (int mi = 0; mi < 4; ++mi) {
        v16h af = load_a_frag(attrS, 256, mi * 16, k0, lane);
        acc[mi][ni] = wmma_f16(af, bf, acc[mi][ni]);
      }
    }
  }
  for (int k0 = 0; k0 < 1280; k0 += 32) {        // spline term, K=256*5
    __syncthreads();
#pragma unroll
    for (int t = 0; t < 8; ++t) {
      int idx = tid + t * 256;
      int n = idx >> 3, c4 = (idx & 7) << 2;
      float4 v = *(const float4*)(Sn + (size_t)n * 1280 + k0 + c4);
      Bs[(c4 + 0) * 256 + n] = (_Float16)v.x;
      Bs[(c4 + 1) * 256 + n] = (_Float16)v.y;
      Bs[(c4 + 2) * 256 + n] = (_Float16)v.z;
      Bs[(c4 + 3) * 256 + n] = (_Float16)v.w;
    }
    __syncthreads();
#pragma unroll
    for (int ni = 0; ni < 2; ++ni) {
      v16h bf = load_b_frag(Bs, 256, w * 32 + ni * 16, lane);
#pragma unroll
      for (int mi = 0; mi < 4; ++mi) {
        v16h af = load_a_frag(basisS, 1280, mi * 16, k0, lane);
        acc[mi][ni] = wmma_f16(af, bf, acc[mi][ni]);
      }
    }
  }
  // epilogue: bias + GELU -> outS, then fused LayerNorm + residual
  const int mo = (lane >> 4) << 3;
#pragma unroll
  for (int mi = 0; mi < 4; ++mi)
#pragma unroll
    for (int ni = 0; ni < 2; ++ni) {
      int n = w * 32 + ni * 16 + (lane & 15);
      float bv = bn[n];
#pragma unroll
      for (int r = 0; r < 8; ++r) {
        int m = mi * 16 + mo + r;
        outS[m * 256 + n] = gelu_exact(acc[mi][ni][r] + bv);
      }
    }
  __syncthreads();
  {
    int row = tid >> 2, q = tid & 3;
    const float* op = outS + row * 256 + q * 64;
    float s = 0.f, ss = 0.f;
    for (int j = 0; j < 64; ++j) { float v = op[j]; s += v; ss += v * v; }
    redS[row][q] = s; redQ[row][q] = ss;
  }
  __syncthreads();
  const float lamv = lam[0];
  for (int t = 0; t < 64; ++t) {
    int idx = tid + t * 256;
    int r2 = idx >> 8, col = idx & 255;
    float m  = (redS[r2][0] + redS[r2][1] + redS[r2][2] + redS[r2][3]) *
               (1.f / 256.f);
    float q2 = (redQ[r2][0] + redQ[r2][1] + redQ[r2][2] + redQ[r2][3]) *
               (1.f / 256.f);
    float var = q2 - m * m;
    float o = (outS[r2 * 256 + col] - m) * rsqrtf(var + 1e-5f) * lng[col] +
              lnb[col];
    float hn = o + lamv * resb[(n0 + r2) * HID + col];
    hout[(n0 + r2) * HID + col] = hn;
    feat[(n0 + r2) * HID + col] = hn;
  }
}

// ---------------------------------------------------------------------------
// Per-scale graph means: Gm[b*3+l][d] = mean over nodes of feats[l].
// ---------------------------------------------------------------------------
__global__ void graph_mean_kernel(const float* __restrict__ f0,
                                  const float* __restrict__ f1,
                                  const float* __restrict__ f2,
                                  float* __restrict__ Gm) {
  int id = blockIdx.x;                 // 0..191 : b = id/3, l = id%3
  int b = id / 3, l = id % 3, d = threadIdx.x;
  const float* f = (l == 0) ? f0 : (l == 1) ? f1 : f2;
  float s = 0.f;
  for (int i = 0; i < NPER; ++i) s += f[((size_t)b * NPER + i) * HID + d];
  Gm[(size_t)id * HID + d] = s * (1.f / (float)NPER);
}

// ---------------------------------------------------------------------------
// Multi-scale fusion attention: 3x3 attention per graph -> scale weights.
// ---------------------------------------------------------------------------
__global__ void fuse_attn_kernel(const float* __restrict__ Qf,
                                 const float* __restrict__ Kf,
                                 const float* __restrict__ Vf,
                                 float* __restrict__ sw) {
  int b = blockIdx.x, d = threadIdx.x;             // 64 threads
  __shared__ float red[64];
  float qv[3], kv[3], vv[3];
#pragma unroll
  for (int l = 0; l < 3; ++l) {
    int row = b * 3 + l;
    qv[l] = Qf[row * ATTDIM + d];
    kv[l] = Kf[row * ATTDIM + d];
    vv[l] = Vf[row * ATTDIM + d];
  }
  float A[3][3];
  for (int l = 0; l < 3; ++l)
    for (int m = 0; m < 3; ++m) {
      red[d] = qv[l] * kv[m];
      __syncthreads();
      for (int o = 32; o > 0; o >>= 1) {
        if (d < o) red[d] += red[d + o];
        __syncthreads();
      }
      A[l][m] = red[0] * 0.125f;                   // / sqrt(64)
      __syncthreads();
    }
#pragma unroll
  for (int l = 0; l < 3; ++l) {
    float mx = fmaxf(A[l][0], fmaxf(A[l][1], A[l][2]));
    float e0 = expf(A[l][0] - mx), e1 = expf(A[l][1] - mx),
          e2 = expf(A[l][2] - mx);
    float s = e0 + e1 + e2;
    A[l][0] = e0 / s; A[l][1] = e1 / s; A[l][2] = e2 / s;
  }
  float Ga[3];
#pragma unroll
  for (int l = 0; l < 3; ++l)
    Ga[l] = A[l][0] * vv[0] + A[l][1] * vv[1] + A[l][2] * vv[2];
  float mx = fmaxf(Ga[0], fmaxf(Ga[1], Ga[2]));
  float e[3], s = 0.f;
#pragma unroll
  for (int l = 0; l < 3; ++l) { e[l] = expf(Ga[l] - mx); s += e[l]; }
#pragma unroll
  for (int l = 0; l < 3; ++l) {
    red[d] = e[l] / s;
    __syncthreads();
    for (int o = 32; o > 0; o >>= 1) {
      if (d < o) red[d] += red[d + o];
      __syncthreads();
    }
    if (d == 0) sw[b * 3 + l] = red[0] * (1.f / 64.f);
    __syncthreads();
  }
}

__global__ void scale_combine_kernel(const float* __restrict__ f0,
                                     const float* __restrict__ f1,
                                     const float* __restrict__ f2,
                                     const float* __restrict__ sw,
                                     float* __restrict__ fused) {
  size_t n = blockIdx.x; int t = threadIdx.x;
  int g = (int)(n >> 9);
  float w0 = sw[g * 3 + 0], w1 = sw[g * 3 + 1], w2 = sw[g * 3 + 2];
  size_t off = n * HID + t;
  fused[off] = w0 * f0[off] + w1 * f1[off] + w2 * f2[off];
}

// ---------------------------------------------------------------------------
// Node MHA, one block per (graph, head). K/V f16 tiles + 8 wave-private
// 16x512 f16 score strips in 192KB LDS; QK^T and PV on WMMA.
// ---------------------------------------------------------------------------
__global__ void mha_attn_kernel(const float* __restrict__ q,
                                const float* __restrict__ k,
                                const float* __restrict__ v,
                                float* __restrict__ atto) {
  extern __shared__ char smem_raw[];
  _Float16* Kh  = (_Float16*)smem_raw;       // [512][32]
  _Float16* Vh  = Kh + NPER * HDIM;          // [512][32]
  _Float16* scS = Vh + NPER * HDIM;          // [8][16][512]
  const int tid = threadIdx.x, lane = tid & 31, w = tid >> 5;
  const int b = blockIdx.x >> 3, head = blockIdx.x & 7;
  const int hoff = head * HDIM;
  const size_t base = (size_t)b * NPER;

#pragma unroll 4
  for (int t = 0; t < 64; ++t) {             // stage K,V (f16)
    int idx = tid + t * 256;                 // 0..16383
    int key = idx >> 5, d = idx & 31;
    size_t g = (base + key) * HID + hoff + d;
    Kh[idx] = (_Float16)k[g];
    Vh[idx] = (_Float16)v[g];
  }
  __syncthreads();

  _Float16* scW = scS + w * (16 * NPER);
  for (int it = 0; it < 4; ++it) {           // 32 q-tiles / 8 waves
    int qt = w + it * 8;
    // A fragment of Q straight from global (f32 -> f16)
    int m = lane & 15, kb = (lane >> 4) << 3;
    const float* qp = q + (base + qt * 16 + m) * HID + hoff + kb;
    v16h aq;
#pragma unroll
    for (int j = 0; j < 8; ++j) {
      aq[j] = (_Float16)qp[j];
      aq[j + 8] = (_Float16)qp[j + 16];
    }
    // scores = Q K^T / sqrt(32)
    for (int kt = 0; kt < 32; ++kt) {
      v16h bk;
#pragma unroll
      for (int j = 0; j < 8; ++j) {
        bk[j] = Kh[(kt * 16 + (lane & 15)) * HDIM + kb + j];
        bk[j + 8] = Kh[(kt * 16 + (lane & 15)) * HDIM + kb + 16 + j];
      }
      v8f sc = {};
      sc = wmma_f16(aq, bk, sc);
      int n = lane & 15, mo = (lane >> 4) << 3;
#pragma unroll
      for (int r = 0; r < 8; ++r)
        scW[(mo + r) * NPER + kt * 16 + n] =
            (_Float16)(sc[r] * 0.17677669529f);
    }
    // wave-private softmax over 512 keys (lane pairs via shfl_xor 16)
    {
      int row = lane & 15, seg = lane >> 4;
      _Float16* sp = scW + row * NPER + seg * 256;
      float mx = -3.0e38f;
      for (int j = 0; j < 256; ++j) mx = fmaxf(mx, (float)sp[j]);
      mx = fmaxf(mx, __shfl_xor(mx, 16, 32));
      float sum = 0.f;
      for (int j = 0; j < 256; ++j) sum += expf((float)sp[j] - mx);
      sum += __shfl_xor(sum, 16, 32);
      float inv = 1.f / sum;
      for (int j = 0; j < 256; ++j)
        sp[j] = (_Float16)(expf((float)sp[j] - mx) * inv);
    }
    // out = P @ V   (K=512 in 16 WMMA steps, N=32 as two frags)
    v8f o0 = {}, o1 = {};
    for (int ks = 0; ks < 16; ++ks) {
      v16h ap = load_a_frag(scW, NPER, 0, ks * 32, lane);
      v16h b0, b1;
#pragma unroll
      for (int j = 0; j < 8; ++j) {
        b0[j] = Vh[(ks * 32 + kb + j) * HDIM + (lane & 15)];
        b0[j + 8] = Vh[(ks * 32 + kb + 16 + j) * HDIM + (lane & 15)];
        b1[j] = Vh[(ks * 32 + kb + j) * HDIM + 16 + (lane & 15)];
        b1[j + 8] = Vh[(ks * 32 + kb + 16 + j) * HDIM + 16 + (lane & 15)];
      }
      o0 = wmma_f16(ap, b0, o0);
      o1 = wmma_f16(ap, b1, o1);
    }
    int n = lane & 15, mo = (lane >> 4) << 3;
#pragma unroll
    for (int r = 0; r < 8; ++r) {
      size_t rowg = (base + qt * 16 + mo + r) * HID + hoff;
      atto[rowg + n] = o0[r];
      atto[rowg + 16 + n] = o1[r];
    }
  }
}

// ---------------------------------------------------------------------------
// nd = LN(fused + att_proj); out = LN(fused + nd). One row per block.
// ---------------------------------------------------------------------------
__device__ __forceinline__ float block_sum256(float v, float* red) {
  int t = threadIdx.x;
  red[t] = v;
  __syncthreads();
  for (int o = 128; o > 0; o >>= 1) {
    if (t < o) red[t] += red[t + o];
    __syncthreads();
  }
  float r = red[0];
  __syncthreads();
  return r;
}

__global__ void residual_ln2_kernel(const float* __restrict__ fused,
                                    const float* __restrict__ proj,
                                    const float* __restrict__ g1,
                                    const float* __restrict__ b1,
                                    const float* __restrict__ g2,
                                    const float* __restrict__ b2,
                                    float* __restrict__ outn) {
  __shared__ float red[256];
  size_t row = blockIdx.x; int t = threadIdx.x;
  float f = fused[row * HID + t];
  float a = f + proj[row * HID + t];
  float m = block_sum256(a, red) * (1.f / 256.f);
  float var = block_sum256((a - m) * (a - m), red) * (1.f / 256.f);
  float nd = (a - m) * rsqrtf(var + 1e-5f) * g1[t] + b1[t];
  float c = f + nd;
  float m2 = block_sum256(c, red) * (1.f / 256.f);
  float v2 = block_sum256((c - m2) * (c - m2), red) * (1.f / 256.f);
  outn[row * HID + t] = (c - m2) * rsqrtf(v2 + 1e-5f) * g2[t] + b2[t];
}

__global__ void pool_kernel(const float* __restrict__ outn,
                            float* __restrict__ gr) {
  int b = blockIdx.x, d = threadIdx.x;
  float s = 0.f;
  for (int i = 0; i < NPER; ++i) s += outn[((size_t)b * NPER + i) * HID + d];
  gr[(size_t)b * HID + d] = s * (1.f / (float)NPER);
}

// ---------------------------------------------------------------------------
// Host-side orchestration.
// ---------------------------------------------------------------------------
static inline void launch_gemm(const float* A, const float* W, const float* bias,
                               float* C, int M, int N, int K, int epi,
                               hipStream_t s) {
  dim3 grid((M + 127) / 128, (N + 63) / 64);
  size_t lds = (128 * 32 + 32 * 64) * sizeof(_Float16);
  gemm_wmma_kernel<<<grid, 256, lds, s>>>(A, W, bias, C, M, N, K, epi);
}

extern "C" void kernel_launch(void* const* d_in, const int* in_sizes, int n_in,
                              void* d_out, int out_size, void* d_ws,
                              size_t ws_size, hipStream_t stream) {
  (void)in_sizes; (void)n_in; (void)out_size; (void)ws_size;
#define FP(i) ((const float*)d_in[(i)])
  const float* x     = FP(0);
  const int*   eix   = (const int*)d_in[1];      // [2, E] (int32)
  const float* eattr = FP(2);                    // [E, 64]
  /* d_in[3] = batch: graphs are equal-size -> index arithmetic instead */
  int p = 4;
  // params pytree, dict keys sorted / tuples positional:
  // cls {W1,W2,b1,b2}
  const float* clsW1 = FP(p++); const float* clsW2 = FP(p++);
  const float* clsb1 = FP(p++); const float* clsb2 = FP(p++);
  // fuse {Wk,Wq,Wv,bk,bq,bv}
  const float* fWk = FP(p++); const float* fWq = FP(p++);
  const float* fWv = FP(p++);
  const float* fbk = FP(p++); const float* fbq = FP(p++);
  const float* fbv = FP(p++);
  // in_proj (W, b)
  const float* Win = FP(p++); const float* bin = FP(p++);
  // layers x3: kan_edge{S,W,b,g}, kan_node{S,W,b,g}, lam, ln(g,b), res(W,b)
  struct Lay {
    const float *eS, *eW, *eb, *eg, *nS, *nW, *nb, *ng, *lam, *lng, *lnb,
        *rW, *rb;
  } L[3];
  for (int l = 0; l < 3; ++l) {
    L[l].eS = FP(p++); L[l].eW = FP(p++); L[l].eb = FP(p++); L[l].eg = FP(p++);
    L[l].nS = FP(p++); L[l].nW = FP(p++); L[l].nb = FP(p++); L[l].ng = FP(p++);
    L[l].lam = FP(p++);
    L[l].lng = FP(p++); L[l].lnb = FP(p++);
    L[l].rW = FP(p++); L[l].rb = FP(p++);
  }
  // ln_out (g, b)
  const float* loG = FP(p++); const float* loB = FP(p++);
  // mha {Wk,Wo,Wq,Wv,bk,bo,bq,bv,ln_b,ln_g}
  const float* mWk = FP(p++); const float* mWo = FP(p++);
  const float* mWq = FP(p++); const float* mWv = FP(p++);
  const float* mbk = FP(p++); const float* mbo = FP(p++);
  const float* mbq = FP(p++); const float* mbv = FP(p++);
  const float* mlnb = FP(p++); const float* mlng = FP(p++);
#undef FP
  const int* esrc = eix;
  const int* edst = eix + NEDGES;

  // ---- workspace carve (fp32) ----
  float* ws = (float*)d_ws;
  size_t off = 0;
  auto alloc = [&](size_t n) { float* r = ws + off; off += n; return r; };
  const size_t ND = (size_t)NNODES * HID;
  float* h      = alloc(ND);
  float* f0     = alloc(ND);
  float* f1     = alloc(ND);
  float* f2     = alloc(ND);
  float* resbuf = alloc(ND);
  float* agg    = alloc(ND);
  float* Gm     = alloc((size_t)NGRAPHS * 3 * HID);
  float* Qf     = alloc((size_t)NGRAPHS * 3 * ATTDIM);
  float* Kf     = alloc((size_t)NGRAPHS * 3 * ATTDIM);
  float* Vf     = alloc((size_t)NGRAPHS * 3 * ATTDIM);
  float* swb    = alloc((size_t)NGRAPHS * 4);   // 3 used, keep 16B aligned
  float* fused  = alloc(ND);
  float* qb     = alloc(ND);
  float* kb     = alloc(ND);
  float* vb     = alloc(ND);
  float* atto   = alloc(ND);
  float* proj   = alloc(ND);
  float* outn   = alloc(ND);
  float* gr     = alloc((size_t)NGRAPHS * HID);
  float* hcls   = alloc((size_t)NGRAPHS * (HID / 2));

  const size_t EDGE_LDS = (size_t)(64 * 64 + 64 * 320 + 32 * 256) * 2;
  const size_t NODE_LDS =
      (size_t)(64 * 256 + 64 * 1280 + 32 * 256) * 2 + (size_t)64 * 256 * 4;
  const size_t ATTN_LDS = (size_t)(NPER * HDIM * 2 + 8 * 16 * NPER) * 2;

  // 1) input projection
  launch_gemm(x, Win, bin, h, NNODES, HID, INDIM, EPI_NONE, stream);

  // 2) three KAN-GNN layers
  float* feats[3] = {f0, f1, f2};
  for (int l = 0; l < 3; ++l) {
    launch_gemm(h, L[l].rW, L[l].rb, resbuf, NNODES, HID, HID, EPI_NONE,
                stream);
    hipMemsetAsync(agg, 0, ND * sizeof(float), stream);
    kan_edge_agg_kernel<<<NEDGES / 64, 256, EDGE_LDS, stream>>>(
        eattr, esrc, edst, h, L[l].eW, L[l].eS, L[l].eb, L[l].eg, agg);
    kan_node_kernel<<<NNODES / 64, 256, NODE_LDS, stream>>>(
        agg, L[l].nW, L[l].nS, L[l].nb, L[l].ng, L[l].lng, L[l].lnb, resbuf,
        L[l].lam, h, feats[l]);
  }

  // 3) multi-scale fusion
  graph_mean_kernel<<<NGRAPHS * 3, HID, 0, stream>>>(f0, f1, f2, Gm);
  launch_gemm(Gm, fWq, fbq, Qf, NGRAPHS * 3, ATTDIM, HID, EPI_NONE, stream);
  launch_gemm(Gm, fWk, fbk, Kf, NGRAPHS * 3, ATTDIM, HID, EPI_NONE, stream);
  launch_gemm(Gm, fWv, fbv, Vf, NGRAPHS * 3, ATTDIM, HID, EPI_NONE, stream);
  fuse_attn_kernel<<<NGRAPHS, 64, 0, stream>>>(Qf, Kf, Vf, swb);
  scale_combine_kernel<<<NNODES, HID, 0, stream>>>(f0, f1, f2, swb, fused);

  // 4) node-level multi-head attention
  launch_gemm(fused, mWq, mbq, qb, NNODES, HID, HID, EPI_NONE, stream);
  launch_gemm(fused, mWk, mbk, kb, NNODES, HID, HID, EPI_NONE, stream);
  launch_gemm(fused, mWv, mbv, vb, NNODES, HID, HID, EPI_NONE, stream);
  mha_attn_kernel<<<NGRAPHS * HEADS, 256, ATTN_LDS, stream>>>(qb, kb, vb,
                                                              atto);
  launch_gemm(atto, mWo, mbo, proj, NNODES, HID, HID, EPI_NONE, stream);
  residual_ln2_kernel<<<NNODES, HID, 0, stream>>>(fused, proj, mlng, mlnb, loG,
                                                  loB, outn);

  // 5) pool + classifier
  pool_kernel<<<NGRAPHS, HID, 0, stream>>>(outn, gr);
  launch_gemm(gr, clsW1, clsb1, hcls, NGRAPHS, HID / 2, HID, EPI_RELU, stream);
  launch_gemm(hcls, clsW2, clsb2, (float*)d_out, NGRAPHS, OUTDIM, HID / 2,
              EPI_NONE, stream);
}
